// EfficientMultiHeadAttention_78219944395546
// MI455X (gfx1250) — matrix operand order repaired and checked by reading, not compiled, not run on hardware
//
#include <hip/hip_runtime.h>
#include <hip/hip_bf16.h>
#include <cstddef>

// Problem constants (match reference)
#define B_   8
#define N_   2048
#define D_   512
#define H_   8
#define DH_  64
#define NK_  1024        // N / SR
#define BH_  64          // B * H

typedef __attribute__((ext_vector_type(16))) __bf16 v16bf;
typedef __attribute__((ext_vector_type(8)))  float  v8f;

union Frag16 { unsigned int u[8]; unsigned short h[16]; v16bf v; };

// Native f32 -> bf16 (RNE fptrunc -> hardware cvt on gfx1250)
__device__ __forceinline__ unsigned short f2bf(float f) {
  return __builtin_bit_cast(unsigned short, static_cast<__bf16>(f));
}
__device__ __forceinline__ unsigned int pack2(float lo, float hi) {
  unsigned int a = __builtin_bit_cast(unsigned short, static_cast<__bf16>(lo));
  unsigned int b = __builtin_bit_cast(unsigned short, static_cast<__bf16>(hi));
  return a | (b << 16);
}

// ---- CDNA5 async global->LDS copy (ASYNCcnt-tracked), 16B per lane ----
// Generic pointers to LDS carry the LDS byte offset in addr[31:0] (ISA 10.2),
// so truncation yields the VDST operand the instruction expects.
__device__ __forceinline__ void async_copy_b128(const void* gsrc, void* ldst) {
  unsigned int lofs = (unsigned int)(unsigned long long)ldst;
  unsigned long long ga = (unsigned long long)gsrc;
  asm volatile("global_load_async_to_lds_b128 %0, %1, off" :: "v"(lofs), "v"(ga) : "memory");
}
__device__ __forceinline__ void wait_asynccnt0() {
  asm volatile("s_wait_asynccnt 0x0" ::: "memory");
}

// ---- WMMA fragment loaders (wave32, 16-bit layouts per CDNA5 ISA 7.12.2) ----
// A (16x32 bf16): lane 0-15 -> row=lane, K = kb..kb+7 (elems 0-7), kb+16..kb+23 (elems 8-15), kb=k0
//                 lane 16-31 -> row=lane-16, kb=k0+8
__device__ __forceinline__ v16bf load_a_bf16(const unsigned short* A, int lda, int row0, int k0) {
  int lane = threadIdx.x & 31;
  const unsigned short* p = A + (row0 + (lane & 15)) * lda + k0 + ((lane >> 4) << 3);
  Frag16 f;
#pragma unroll
  for (int j = 0; j < 4; ++j) {
    f.u[j]     = *(const unsigned int*)(p + 2 * j);
    f.u[4 + j] = *(const unsigned int*)(p + 16 + 2 * j);
  }
  return f.v;
}
// Same A layout but source is fp32 (converted on the fly)
__device__ __forceinline__ v16bf load_a_f32(const float* A, int lda, int row0, int k0) {
  int lane = threadIdx.x & 31;
  const float* p = A + (row0 + (lane & 15)) * lda + k0 + ((lane >> 4) << 3);
  Frag16 f;
#pragma unroll
  for (int j = 0; j < 4; ++j) {
    f.u[j]     = pack2(p[2 * j], p[2 * j + 1]);
    f.u[4 + j] = pack2(p[16 + 2 * j], p[17 + 2 * j]);
  }
  return f.v;
}
// B (32x16 bf16) from K-transposed storage Bt[n][k]: lane 0-15 -> col=lane, K=kb..kb+7 (elems 0-7),
// kb+16..kb+23 (elems 8-15), kb=k0; lane 16-31 -> col=lane-16, kb=k0+8 (same K order as A).
__device__ __forceinline__ v16bf load_b_tn(const unsigned short* Bt, int ldb, int n0, int k0) {
  int lane = threadIdx.x & 31;
  const unsigned short* p = Bt + (n0 + (lane & 15)) * ldb + k0 + ((lane >> 4) << 3);
  Frag16 f;
#pragma unroll
  for (int j = 0; j < 4; ++j) {
    f.u[j]     = *(const unsigned int*)(p + 2 * j);
    f.u[4 + j] = *(const unsigned int*)(p + 16 + 2 * j);
  }
  return f.v;
}

#define WMMA_BF16(a, b, c) \
  __builtin_amdgcn_wmma_f32_16x16x32_bf16(false, (a), false, (b), (short)0, (c), false, false)

// ============================================================================
// Kernel 1: Q = x @ Wq + bq, stored bf16 in head-major layout Qbuf[h*B+b][n][dh]
// 64x64 tile / block, 8 waves, 2 WMMA subtiles per wave.
// ============================================================================
__global__ __launch_bounds__(256) void k_qproj(const float* __restrict__ x,
                                               const float* __restrict__ Wq,
                                               const float* __restrict__ bq,
                                               unsigned short* __restrict__ Qbuf) {
  __shared__ unsigned short As[64 * 32];
  __shared__ unsigned short Wt[64 * 32];  // transposed [col][k]
  int tid = threadIdx.x, wave = tid >> 5, lane = tid & 31;
  int Rb = blockIdx.x * 64, Cb = blockIdx.y * 64;
  v8f acc[2] = {};
  for (int k0 = 0; k0 < D_; k0 += 32) {
#pragma unroll
    for (int j = 0; j < 4; ++j) {
      int e = tid * 8 + j * 2;                 // 2048 halves each
      int r = e >> 5, c = e & 31;
      ((unsigned int*)As)[e >> 1] = pack2(x[(size_t)(Rb + r) * D_ + k0 + c],
                                          x[(size_t)(Rb + r) * D_ + k0 + c + 1]);
      int cc = e >> 5, kk = e & 31;
      ((unsigned int*)Wt)[e >> 1] = pack2(Wq[(size_t)(k0 + kk) * D_ + Cb + cc],
                                          Wq[(size_t)(k0 + kk + 1) * D_ + Cb + cc]);
    }
    __syncthreads();
#pragma unroll
    for (int s = 0; s < 2; ++s) {
      int st = wave * 2 + s, mt = st >> 2, nt = st & 3;
      v16bf a = load_a_bf16(As, 32, mt * 16, 0);
      v16bf b = load_b_tn(Wt, 32, nt * 16, 0);
      acc[s] = WMMA_BF16(a, b, acc[s]);
    }
    __syncthreads();
  }
  int rofs = (lane >> 4) * 8, col = lane & 15;
  for (int s = 0; s < 2; ++s) {
    int st = wave * 2 + s, mt = st >> 2, nt = st & 3;
    int gc = Cb + nt * 16 + col;
    float bias = bq[gc];
    int h = gc >> 6, dh = gc & 63;
#pragma unroll
    for (int i = 0; i < 8; ++i) {
      int gr = Rb + mt * 16 + rofs + i;
      int bb = gr >> 11, n = gr & (N_ - 1);
      Qbuf[(size_t)((h * B_ + bb) * N_ + n) * DH_ + dh] = f2bf(acc[s][i] + bias);
    }
  }
}

// ============================================================================
// Kernel 2: spatial-reduction conv as GEMM: xk = Xkv(8192x1024) @ W'(1024x512)+bsr
// W'[k][o] = Wsr[o, k%512, k/512]  (Wsr is (D_out, D_in, SR) row-major)
// ============================================================================
__global__ __launch_bounds__(256) void k_srconv(const float* __restrict__ xkv,
                                                const float* __restrict__ Wsr,
                                                const float* __restrict__ bsr,
                                                float* __restrict__ xk) {
  __shared__ unsigned short As[64 * 32];
  __shared__ unsigned short Wt[64 * 32];
  int tid = threadIdx.x, wave = tid >> 5, lane = tid & 31;
  int Rb = blockIdx.x * 64, Cb = blockIdx.y * 64;
  const int K = 2 * D_;  // 1024
  v8f acc[2] = {};
  for (int k0 = 0; k0 < K; k0 += 32) {
#pragma unroll
    for (int j = 0; j < 4; ++j) {
      int e = tid * 8 + j * 2;
      int r = e >> 5, c = e & 31;
      ((unsigned int*)As)[e >> 1] = pack2(xkv[(size_t)(Rb + r) * K + k0 + c],
                                          xkv[(size_t)(Rb + r) * K + k0 + c + 1]);
      int cc = e >> 5, kk = e & 31;
      int kg0 = k0 + kk, kg1 = kg0 + 1;
      float w0 = Wsr[(size_t)(Cb + cc) * K + (kg0 & 511) * 2 + (kg0 >> 9)];
      float w1 = Wsr[(size_t)(Cb + cc) * K + (kg1 & 511) * 2 + (kg1 >> 9)];
      ((unsigned int*)Wt)[e >> 1] = pack2(w0, w1);
    }
    __syncthreads();
#pragma unroll
    for (int s = 0; s < 2; ++s) {
      int st = wave * 2 + s, mt = st >> 2, nt = st & 3;
      v16bf a = load_a_bf16(As, 32, mt * 16, 0);
      v16bf b = load_b_tn(Wt, 32, nt * 16, 0);
      acc[s] = WMMA_BF16(a, b, acc[s]);
    }
    __syncthreads();
  }
  int rofs = (lane >> 4) * 8, col = lane & 15;
  for (int s = 0; s < 2; ++s) {
    int st = wave * 2 + s, mt = st >> 2, nt = st & 3;
    int gc = Cb + nt * 16 + col;
    float bias = bsr[gc];
#pragma unroll
    for (int i = 0; i < 8; ++i)
      xk[(size_t)(Rb + mt * 16 + rofs + i) * D_ + gc] = acc[s][i] + bias;
  }
}

// ============================================================================
// Kernel 3: LayerNorm over D=512 per row, emit bf16
// ============================================================================
__global__ __launch_bounds__(256) void k_layernorm(const float* __restrict__ xk,
                                                   const float* __restrict__ gamma,
                                                   const float* __restrict__ beta,
                                                   unsigned short* __restrict__ xkbf) {
  __shared__ float red[256];
  int tid = threadIdx.x;
  size_t base = (size_t)blockIdx.x * D_;
  float v0 = xk[base + tid], v1 = xk[base + tid + 256];
  red[tid] = v0 + v1;
  __syncthreads();
  for (int o = 128; o > 0; o >>= 1) { if (tid < o) red[tid] += red[tid + o]; __syncthreads(); }
  float mu = red[0] * (1.0f / D_);
  __syncthreads();
  float d0 = v0 - mu, d1 = v1 - mu;
  red[tid] = d0 * d0 + d1 * d1;
  __syncthreads();
  for (int o = 128; o > 0; o >>= 1) { if (tid < o) red[tid] += red[tid + o]; __syncthreads(); }
  float rstd = rsqrtf(red[0] * (1.0f / D_) + 1e-5f);
  xkbf[base + tid]       = f2bf(d0 * rstd * gamma[tid] + beta[tid]);
  xkbf[base + tid + 256] = f2bf(d1 * rstd * gamma[tid + 256] + beta[tid + 256]);
}

// ============================================================================
// Kernel 4: KV = xk @ Wkv + bkv, scattered bf16 into K/V with the reference's
// "faithful reshape": row r -> (b = r/1024, t = r%1024); bh = t>>4,
// m = (t&15)*64 + b*8 + h.  Kbuf/Vbuf layout: [bh][m][dh].
// A-tile staged with CDNA5 async global->LDS copies (overlapped with W convert).
// ============================================================================
__global__ __launch_bounds__(256) void k_kvproj(const unsigned short* __restrict__ xkbf,
                                                const float* __restrict__ Wkv,
                                                const float* __restrict__ bkv,
                                                unsigned short* __restrict__ Kbuf,
                                                unsigned short* __restrict__ Vbuf) {
  __shared__ unsigned short As[64 * 32];
  __shared__ unsigned short Wt[64 * 32];
  int tid = threadIdx.x, wave = tid >> 5, lane = tid & 31;
  int Rb = blockIdx.x * 64, Cb = blockIdx.y * 64;
  v8f acc[2] = {};
  for (int k0 = 0; k0 < D_; k0 += 32) {
    // async stage A (bf16, straight copy): 64 rows x 32 halves = 4 KB = 256 lanes * 16 B
    async_copy_b128(xkbf + (size_t)(Rb + (tid >> 2)) * D_ + k0 + (tid & 3) * 8,
                    As + tid * 8);
    // convert W chunk while the async copy is in flight
#pragma unroll
    for (int j = 0; j < 4; ++j) {
      int e2 = tid * 8 + j * 2;                // 2048 halves
      int cc = e2 >> 5, kk = e2 & 31;
      ((unsigned int*)Wt)[e2 >> 1] = pack2(Wkv[(size_t)(k0 + kk) * (2 * D_) + Cb + cc],
                                           Wkv[(size_t)(k0 + kk + 1) * (2 * D_) + Cb + cc]);
    }
    wait_asynccnt0();
    __syncthreads();
#pragma unroll
    for (int s = 0; s < 2; ++s) {
      int st = wave * 2 + s, mt = st >> 2, nt = st & 3;
      v16bf a = load_a_bf16(As, 32, mt * 16, 0);
      v16bf b = load_b_tn(Wt, 32, nt * 16, 0);
      acc[s] = WMMA_BF16(a, b, acc[s]);
    }
    __syncthreads();
  }
  int rofs = (lane >> 4) * 8, col = lane & 15;
  for (int s = 0; s < 2; ++s) {
    int st = wave * 2 + s, mt = st >> 2, nt = st & 3;
    int gc = Cb + nt * 16 + col;
    float bias = bkv[gc];
    int isV = gc >> 9, c = gc & 511;
    int h = c >> 6, dh = c & 63;
    unsigned short* dst = isV ? Vbuf : Kbuf;
#pragma unroll
    for (int i = 0; i < 8; ++i) {
      int gr = Rb + mt * 16 + rofs + i;
      int bb = gr >> 10, t = gr & 1023;
      int bh = t >> 4, mm = ((t & 15) << 6) + bb * 8 + h;
      dst[(size_t)(bh * NK_ + mm) * DH_ + dh] = f2bf(acc[s][i] + bias);
    }
  }
}

// ============================================================================
// Kernel 5: attention. One block = one (bh, 64-row) tile.
// Full 64x1024 fp32 score tile lives in LDS (256 KB of the WGP's 320 KB):
//   S = (Q K^T)/8 via WMMA  ->  softmax in place  ->  stream probs to d_out
//   ->  ctx = P @ V via WMMA (P converted bf16 per-fragment from LDS).
// Q tile staged with async global->LDS.
// ============================================================================
__global__ __launch_bounds__(256) void k_attn(const unsigned short* __restrict__ Qbuf,
                                              const unsigned short* __restrict__ Kbuf,
                                              const unsigned short* __restrict__ Vbuf,
                                              float* __restrict__ attn_out,
                                              float* __restrict__ ctxws) {
  extern __shared__ unsigned char smem[];
  float* S            = (float*)smem;                                  // 64*1024 f32
  unsigned short* Qs  = (unsigned short*)(smem + 262144);              // 64*64 bf16
  unsigned short* Vt  = (unsigned short*)(smem + 262144 + 8192);       // 64(dh) x 32(k) bf16
  float* red          = (float*)(smem + 262144 + 8192 + 4096);         // 256 f32

  int tid = threadIdx.x, wave = tid >> 5, lane = tid & 31;
  int bh = blockIdx.x, n0 = blockIdx.y * 64;
  const unsigned short* Qg = Qbuf + (size_t)(bh * N_ + n0) * DH_;
  const unsigned short* Kg = Kbuf + (size_t)bh * NK_ * DH_;
  const unsigned short* Vg = Vbuf + (size_t)bh * NK_ * DH_;

  // async stage Q tile: 8 KB = 2 rounds of 256 lanes * 16 B
  async_copy_b128(Qg + tid * 8,        Qs + tid * 8);
  async_copy_b128(Qg + 2048 + tid * 8, Qs + 2048 + tid * 8);
  wait_asynccnt0();
  __syncthreads();

  int rofs = (lane >> 4) * 8, col = lane & 15;

  // --- S = Q K^T * (1/sqrt(DH)); K rows are exactly B-transposed layout ---
  for (int s = 0; s < 32; ++s) {
    int mt = s & 3;
    int jt = wave * 8 + (s >> 2);
    __builtin_prefetch(Kg + (size_t)((jt + 1) & 63) * 16 * DH_ + lane * 4, 0, 1);
    v8f acc = {};
#pragma unroll
    for (int kk = 0; kk < 64; kk += 32) {
      v16bf a = load_a_bf16(Qs, 64, mt * 16, kk);
      v16bf b = load_b_tn(Kg, 64, jt * 16, kk);     // straight from global (2x16B per lane)
      acc = WMMA_BF16(a, b, acc);
    }
#pragma unroll
    for (int i = 0; i < 8; ++i)
      S[(mt * 16 + rofs + i) * 1024 + jt * 16 + col] = acc[i] * 0.125f;
  }
  __syncthreads();

  // --- softmax: 4 threads per row, 256 cols each ---
  int row = tid >> 2, part = tid & 3;
  float* Sr = S + row * 1024 + part * 256;
  float m = -3.402823e38f;
  for (int j = 0; j < 256; ++j) m = fmaxf(m, Sr[j]);
  red[tid] = m;
  __syncthreads();
  float rm = fmaxf(fmaxf(red[row * 4], red[row * 4 + 1]),
                   fmaxf(red[row * 4 + 2], red[row * 4 + 3]));
  __syncthreads();
  float sum = 0.f;
  for (int j = 0; j < 256; ++j) { float e = __expf(Sr[j] - rm); Sr[j] = e; sum += e; }
  red[tid] = sum;
  __syncthreads();
  float inv = 1.f / (red[row * 4] + red[row * 4 + 1] + red[row * 4 + 2] + red[row * 4 + 3]);
  float* Ao = attn_out + ((size_t)bh * N_ + n0 + row) * NK_ + part * 256;
  for (int j = 0; j < 256; ++j) { float p = Sr[j] * inv; Sr[j] = p; Ao[j] = p; }
  __syncthreads();

  // --- ctx = P @ V; stage V^T chunk (32 k-rows) in LDS per step ---
  v8f cacc[2] = {};
  for (int k0 = 0; k0 < NK_; k0 += 32) {
#pragma unroll
    for (int j = 0; j < 4; ++j) {
      int e = tid * 4 + j;                      // 1024 dwords = 64 dh x 16
      int dh = e >> 4, kk2 = (e & 15) * 2;
      unsigned int lo = Vg[(size_t)(k0 + kk2) * DH_ + dh];
      unsigned int hi = Vg[(size_t)(k0 + kk2 + 1) * DH_ + dh];
      ((unsigned int*)Vt)[e] = lo | (hi << 16);
    }
    __syncthreads();
#pragma unroll
    for (int s = 0; s < 2; ++s) {
      int st = wave * 2 + s, mt = st >> 2, nt = st & 3;
      v16bf a = load_a_f32(S, 1024, mt * 16, k0);
      v16bf b = load_b_tn(Vt, 32, nt * 16, 0);
      cacc[s] = WMMA_BF16(a, b, cacc[s]);
    }
    __syncthreads();
  }
  for (int s = 0; s < 2; ++s) {
    int st = wave * 2 + s, mt = st >> 2, nt = st & 3;
#pragma unroll
    for (int i = 0; i < 8; ++i)
      ctxws[(size_t)(bh * N_ + n0 + mt * 16 + rofs + i) * DH_ + nt * 16 + col] = cacc[s][i];
  }
}

// ============================================================================
// Kernel 6: out = ctx @ Wp + bp, permuted (H,B,N,DH) -> (B,N,H*DH)
// ============================================================================
__global__ __launch_bounds__(256) void k_outproj(const float* __restrict__ ctxws,
                                                 const float* __restrict__ Wp,
                                                 const float* __restrict__ bp,
                                                 float* __restrict__ out) {
  __shared__ unsigned short As[64 * 64];
  __shared__ unsigned short Wt[64 * 64];     // transposed [dh_out][k]
  int tid = threadIdx.x, wave = tid >> 5, lane = tid & 31;
  int bh = blockIdx.x, n0 = blockIdx.y * 64;
  const float* Ag = ctxws + (size_t)(bh * N_ + n0) * DH_;
#pragma unroll
  for (int j = 0; j < 8; ++j) {
    int e = tid + j * 256;                   // 2048 dwords
    ((unsigned int*)As)[e] = pack2(Ag[2 * e], Ag[2 * e + 1]);
    int c = e >> 5, k2 = (e & 31) * 2;
    ((unsigned int*)Wt)[e] = pack2(Wp[(size_t)k2 * DH_ + c], Wp[(size_t)(k2 + 1) * DH_ + c]);
  }
  __syncthreads();
  v8f acc[2] = {};
  for (int s = 0; s < 2; ++s) {
    int st = wave * 2 + s, mt = st >> 2, nt = st & 3;
#pragma unroll
    for (int kk = 0; kk < 64; kk += 32) {
      v16bf a = load_a_bf16(As, 64, mt * 16, kk);
      v16bf b = load_b_tn(Wt, 64, nt * 16, kk);
      acc[s] = WMMA_BF16(a, b, acc[s]);
    }
  }
  int rofs = (lane >> 4) * 8, col = lane & 15;
  int hh = bh >> 3, bb = bh & 7;
  for (int s = 0; s < 2; ++s) {
    int st = wave * 2 + s, mt = st >> 2, nt = st & 3;
    int c = nt * 16 + col;
    float bias = bp[c];
#pragma unroll
    for (int i = 0; i < 8; ++i) {
      int n = n0 + mt * 16 + rofs + i;
      out[((size_t)(bb * N_ + n) * H_ + hh) * DH_ + c] = acc[s][i] + bias;
    }
  }
}

// ============================================================================
extern "C" void kernel_launch(void* const* d_in, const int* in_sizes, int n_in,
                              void* d_out, int out_size, void* d_ws, size_t ws_size,
                              hipStream_t stream) {
  const float* x     = (const float*)d_in[0];
  const float* x_kv  = (const float*)d_in[1];
  const float* Wq    = (const float*)d_in[2];
  const float* bq    = (const float*)d_in[3];
  const float* Wsr   = (const float*)d_in[4];
  const float* bsr   = (const float*)d_in[5];
  const float* gamma = (const float*)d_in[6];
  const float* beta  = (const float*)d_in[7];
  const float* Wkv   = (const float*)d_in[8];
  const float* bkv   = (const float*)d_in[9];
  const float* Wp    = (const float*)d_in[10];
  const float* bp    = (const float*)d_in[11];

  float* out_ctx  = (float*)d_out;                               // B*N*D
  float* out_attn = out_ctx + (size_t)B_ * N_ * D_;              // BH*N*NK

  // Workspace layout (~88 MB total)
  unsigned char* ws = (unsigned char*)d_ws;
  unsigned short* Qbuf  = (unsigned short*)(ws);                         // 16 MB bf16 [h*B+b][n][dh]
  float*          xk    = (float*)        (ws + ((size_t)16 << 20));     // 16 MB f32  [8192][512]
  unsigned short* xkbf  = (unsigned short*)(ws + ((size_t)32 << 20));    //  8 MB bf16
  unsigned short* Kbuf  = (unsigned short*)(ws + ((size_t)40 << 20));    //  8 MB bf16 [bh][m][dh]
  unsigned short* Vbuf  = (unsigned short*)(ws + ((size_t)48 << 20));    //  8 MB bf16
  float*          ctxws = (float*)        (ws + ((size_t)56 << 20));     // 32 MB f32  [bh][n][dh]

  k_qproj   <<<dim3(256, 8),  256, 0, stream>>>(x, Wq, bq, Qbuf);
  k_srconv  <<<dim3(128, 8),  256, 0, stream>>>(x_kv, Wsr, bsr, xk);
  k_layernorm<<<8192,         256, 0, stream>>>(xk, gamma, beta, xkbf);
  k_kvproj  <<<dim3(128, 16), 256, 0, stream>>>(xkbf, Wkv, bkv, Kbuf, Vbuf);
  // dynamic LDS: 256K (scores) + 8K (Q) + 4K (V^T stage) + 1K (reductions)
  k_attn    <<<dim3(64, 32),  256, 275456, stream>>>(Qbuf, Kbuf, Vbuf, out_attn, ctxws);
  k_outproj <<<dim3(64, 32),  256, 0, stream>>>(ctxws, Wp, bp, out_ctx);
}
